// RefineLabels_26826365731331
// MI455X (gfx1250) — matrix operand order repaired and checked by reading, not compiled, not run on hardware
//
#include <hip/hip_runtime.h>
#include <hip/hip_bf16.h>
#include <math.h>

// Inputs (setup_inputs order):
//  d_in[0] inputs          [256*2048] f32   -- dead (normalize result unused by loss)
//  d_in[1] inputs_logits   [256*1000] f32
//  d_in[2] targets         [256]      i32
//  d_in[3] indexes         [256]      i32
//  d_in[4] neighbors       [13000*13000] bool (1 byte/elem)
//  d_in[5] neighbor_dists  [13000*13000] f32 -- dead
// d_out: 1 float (scalar loss)

typedef __attribute__((ext_vector_type(16))) _Float16 v16h;
typedef __attribute__((ext_vector_type(8)))  _Float16 v8h;
typedef __attribute__((ext_vector_type(8)))  float    v8f;

#define B_    256
#define C_    1000
#define CP    1024      // padded K for WMMA
#define NSAMP 13000
#define TOPK  30

// ---------------------------------------------------------------------------
// Kernel 1: per-row logZ, 30th-largest threshold, masked softmax.
// Emits f16 copies of logits (A) and P (B of the transposed GEMM), K-padded.
// ---------------------------------------------------------------------------
__global__ __launch_bounds__(256) void rl_prep(const float* __restrict__ logits,
                                               _Float16* __restrict__ Af,
                                               _Float16* __restrict__ Pf,
                                               float* __restrict__ logZ)
{
    __shared__ float vals[CP];   // original row (padded with -big)
    __shared__ float work[CP];   // destroyed by top-k extraction
    __shared__ float red[256];
    __shared__ int   redi[256];
    __shared__ float sc[2];
    const int row = blockIdx.x;
    const int t   = threadIdx.x;
    const float NEG = -3.0e38f;

    for (int c = t; c < CP; c += 256) {
        float v = (c < C_) ? logits[row * C_ + c] : NEG;
        vals[c] = v;
        work[c] = v;
    }
    __syncthreads();

    // row max
    float lm = NEG;
    for (int c = t; c < CP; c += 256) lm = fmaxf(lm, vals[c]);
    red[t] = lm; __syncthreads();
    for (int s = 128; s > 0; s >>= 1) { if (t < s) red[t] = fmaxf(red[t], red[t + s]); __syncthreads(); }
    const float rowmax = red[0];
    __syncthreads();

    // log-sum-exp of the raw row (for log_softmax)
    float se = 0.f;
    for (int c = t; c < C_; c += 256) se += __expf(vals[c] - rowmax);
    red[t] = se; __syncthreads();
    for (int s = 128; s > 0; s >>= 1) { if (t < s) red[t] += red[t + s]; __syncthreads(); }
    if (t == 0) logZ[row] = rowmax + __logf(red[0]);
    __syncthreads();

    // 30th-largest value via iterative argmax extraction (exact w/ duplicates)
    float kth = NEG;
    for (int it = 0; it < TOPK; ++it) {
        float m = NEG; int mi = 0;
        for (int c = t; c < CP; c += 256) { float v = work[c]; if (v > m) { m = v; mi = c; } }
        red[t] = m; redi[t] = mi; __syncthreads();
        for (int s = 128; s > 0; s >>= 1) {
            if (t < s && red[t + s] > red[t]) { red[t] = red[t + s]; redi[t] = redi[t + s]; }
            __syncthreads();
        }
        if (t == 0) { sc[0] = red[0]; work[redi[0]] = NEG; }
        __syncthreads();
        kth = sc[0];
    }

    // masked softmax: entries < kth become 0.0 and STILL participate in softmax
    const float mmax = fmaxf(rowmax, 0.0f);
    float sm = 0.f;
    for (int c = t; c < C_; c += 256) {
        float v  = vals[c];
        float mv = (v >= kth) ? v : 0.0f;
        sm += __expf(mv - mmax);
    }
    red[t] = sm; __syncthreads();
    for (int s = 128; s > 0; s >>= 1) { if (t < s) red[t] += red[t + s]; __syncthreads(); }
    const float inv = 1.0f / red[0];

    for (int c = t; c < CP; c += 256) {
        if (c < C_) {
            float v  = vals[c];
            float mv = (v >= kth) ? v : 0.0f;
            Pf[row * CP + c] = (_Float16)(__expf(mv - mmax) * inv);
            Af[row * CP + c] = (_Float16)v;
        } else {
            Pf[row * CP + c] = (_Float16)0.0f;   // zero K-padding: no GEMM contribution
            Af[row * CP + c] = (_Float16)0.0f;
        }
    }
}

// ---------------------------------------------------------------------------
// Kernel 2: in-batch adjacency gather M[i,j] = neighbors[idx_i, idx_j] + row counts
// ---------------------------------------------------------------------------
__global__ __launch_bounds__(256) void rl_gatherM(const unsigned char* __restrict__ neigh,
                                                  const int* __restrict__ idx,
                                                  float* __restrict__ M,
                                                  float* __restrict__ counts)
{
    __shared__ float red[256];
    const int i = blockIdx.x;
    const int j = threadIdx.x;
    const long long base = (long long)idx[i] * (long long)NSAMP;
    const float m = neigh[base + idx[j]] ? 1.0f : 0.0f;
    M[i * B_ + j] = m;
    red[j] = m; __syncthreads();
    for (int s = 128; s > 0; s >>= 1) { if (j < s) red[j] += red[j + s]; __syncthreads(); }
    if (j == 0) counts[i] = red[0];
}

// ---------------------------------------------------------------------------
// Kernel 3: G = logits_f16 [256x1024] x P_f16^T [1024x256] via v_wmma_f32_16x16x32_f16
// One wave per 16x16 output tile; 256 tiles -> 32 blocks x 8 waves.
// ---------------------------------------------------------------------------
__global__ __launch_bounds__(256) void rl_gemm(const _Float16* __restrict__ Af,
                                               const _Float16* __restrict__ Bf,
                                               float* __restrict__ G)
{
    const int lane = threadIdx.x & 31;
    const int wave = threadIdx.x >> 5;
    const int wg   = blockIdx.x * 8 + wave;      // 0..255
    const int tm   = (wg >> 4) << 4;             // tile row base in G
    const int tn   = (wg & 15) << 4;             // tile col base in G
    const int half = lane >> 4;                  // lane group (0: lanes 0-15, 1: 16-31)
    const int l16  = lane & 15;

    // A 16x32 f16 layout: lane L holds M=L%16; halves 0-7 -> K = k+(L/16)*8..+8,
    //                     halves 8-15 -> K = k+16+(L/16)*8..+8   (two b128 loads)
    const _Float16* a0 = Af + (tm + l16) * CP + half * 8;
    // B 32x16 f16 layout (B = P^T, column n of B = row n of P): lane L holds N=L%16;
    //                     halves 0-7 -> K = k+(L/16)*8..+8, halves 8-15 -> K = k+16+(L/16)*8..+8 (two b128 loads)
    const _Float16* b0 = Bf + (tn + l16) * CP + half * 8;

    v8f acc = {};
    for (int k = 0; k < CP; k += 32) {
        __builtin_prefetch(a0 + k + 128, 0, 1);   // global_prefetch_b8
        __builtin_prefetch(b0 + k + 128, 0, 1);
        union { v16h v; v8h h[2]; } au;
        au.h[0] = *(const v8h*)(a0 + k);
        au.h[1] = *(const v8h*)(a0 + k + 16);
        union { v16h v; v8h h[2]; } bu;
        bu.h[0] = *(const v8h*)(b0 + k);
        bu.h[1] = *(const v8h*)(b0 + k + 16);
        v16h bv = bu.v;
        acc = __builtin_amdgcn_wmma_f32_16x16x32_f16(
                  /*neg_a=*/false, au.v, /*neg_b=*/false, bv,
                  /*c_mod=*/(short)0, acc, /*reuse_a=*/false, /*reuse_b=*/false);
    }

    // D layout: VGPR r, lanes 0-15 -> M=r, lanes 16-31 -> M=8+r; N = lane%16
    float* gout = G + (tm + half * 8) * B_ + tn + l16;
#pragma unroll
    for (int r = 0; r < 8; ++r) gout[r * B_] = acc[r];
}

// ---------------------------------------------------------------------------
// Kernel 4: loss = mean_i [ logZ_i - 0.1*logits[i,t_i] - 0.9/counts_i * sum_j M[i,j]*G[i,j] ]
// (uses sum_c refined = 1; sum_c (M@P)_norm = 1 exactly)
// ---------------------------------------------------------------------------
__global__ __launch_bounds__(256) void rl_finalize(const float* __restrict__ logits,
                                                   const int* __restrict__ targets,
                                                   const float* __restrict__ M,
                                                   const float* __restrict__ counts,
                                                   const float* __restrict__ G,
                                                   const float* __restrict__ logZ,
                                                   float* __restrict__ out)
{
    __shared__ float red[256];
    const int i = threadIdx.x;          // one thread per row
    float dot = 0.f;
    for (int j = 0; j < B_; ++j) dot += M[i * B_ + j] * G[i * B_ + j];
    float li = logZ[i] - 0.1f * logits[i * C_ + targets[i]] - 0.9f * dot / counts[i];
    red[i] = li; __syncthreads();
    for (int s = 128; s > 0; s >>= 1) { if (i < s) red[i] += red[i + s]; __syncthreads(); }
    if (i == 0) out[0] = red[0] * (1.0f / 256.0f);
}

// ---------------------------------------------------------------------------
extern "C" void kernel_launch(void* const* d_in, const int* in_sizes, int n_in,
                              void* d_out, int out_size, void* d_ws, size_t ws_size,
                              hipStream_t stream) {
    const float*         logits    = (const float*)d_in[1];
    const int*           targets   = (const int*)d_in[2];
    const int*           indexes   = (const int*)d_in[3];
    const unsigned char* neighbors = (const unsigned char*)d_in[4];  // jax bool: 1 byte/elem
    float* out = (float*)d_out;

    // workspace layout (~1.6 MB)
    char* ws = (char*)d_ws;
    _Float16* Af     = (_Float16*)(ws);                       // 256*1024*2 = 512 KB
    _Float16* Pf     = (_Float16*)(ws + (512 << 10));         // 512 KB
    float*    G      = (float*)   (ws + (1024 << 10));        // 256 KB
    float*    M      = (float*)   (ws + (1280 << 10));        // 256 KB
    float*    counts = (float*)   (ws + (1536 << 10));        // 1 KB
    float*    logZ   = (float*)   (ws + (1537 << 10));        // 1 KB

    rl_prep    <<<B_, 256, 0, stream>>>(logits, Af, Pf, logZ);
    rl_gatherM <<<B_, 256, 0, stream>>>(neighbors, indexes, M, counts);
    rl_gemm    <<<32,  256, 0, stream>>>(Af, Pf, G);
    rl_finalize<<<1,   256, 0, stream>>>(logits, targets, M, counts, G, logZ, out);
}
